// MambaEncoder_27530740368063
// MI455X (gfx1250) — matrix operand order repaired and checked by reading, not compiled, not run on hardware
//
#include <hip/hip_runtime.h>
#include <hip/hip_bf16.h>
#include <math.h>

// ---------------------------------------------------------------------------
// MI455X / gfx1250 Mamba encoder.
// Dense math: v_wmma_f32_16x16x32_bf16 (f32 accumulate). Weights are
// pre-converted to bf16 once and B tiles are staged into LDS by the Tensor
// Data Mover (tensor_load_to_lds + s_wait_tensorcnt), with TDM pad fields
// giving an 80B LDS row stride (bank-conflict-free, 16B aligned) and TDM OOB
// zero-fill handling the ragged N=48 / K=16 edges. conv2 is an implicit
// im2col GEMM (M=60648, N=256, K=2304) with an LDS k->(ci,kh,kw) table.
// ---------------------------------------------------------------------------

typedef __bf16 bf16_t;
typedef __attribute__((ext_vector_type(16))) __bf16 v16bf;
typedef __attribute__((ext_vector_type(8)))  float  v8f;
typedef __attribute__((ext_vector_type(4)))  unsigned int u32x4;
typedef __attribute__((ext_vector_type(8)))  int          i32x8;
typedef __attribute__((ext_vector_type(4)))  int          i32x4;

#define BB   8
#define TT   399
#define DM   256
#define DI   512
#define DS   16
#define DTR  16
#define BT   (BB*TT)          // 3192 rows
#define CONV1_H 799
#define CONV1_W 39
#define CONV2_F 19
#define EMB_K  4864           // 256*19

#define TILE_M 64
#define TILE_N 64
#define TILE_K 32
#define BSTRIDE 40            // bf16 elems per LDS B row (80B: TDM-padded)
#define GEMM_THREADS 128

// ---------------------------------------------------------------------------
// Generic WMMA GEMM:  C[M,N] = act(A[M,K] @ W[N,K]^T + bias), W in bf16.
// AMODE: 0 = A is f32 row-major (lda); 1 = implicit im2col from bf16 conv1 out
// SMODE: 0 = f32 row-major store (ldc); 1 = scatter into (bt, n*19+f) layout
// ACT:   0 = none, 1 = relu, 2 = softplus, 3 = silu
// ---------------------------------------------------------------------------
template <int AMODE, int SMODE, int ACT>
__global__ __launch_bounds__(GEMM_THREADS)
void gemm_wmma(const float* __restrict__ A, int lda,
               const bf16_t* __restrict__ Abf,
               const bf16_t* __restrict__ Wbf, int ldw,
               const float* __restrict__ bias,
               float* __restrict__ C, int ldc,
               int M, int N, int K)
{
    __shared__ __align__(16) bf16_t sA[TILE_M][TILE_K + 2]; // 68B rows (17 banks)
    __shared__ __align__(16) bf16_t sB[TILE_N][BSTRIDE];    // 80B rows, TDM-filled
    __shared__ int sKbase[TILE_K];   // im2col: ci*H*W + kw per k
    __shared__ int sKh[TILE_K];      // im2col: kh-2 per k

    const int tid  = threadIdx.x;
    const int lane = tid & 31;
    const int wave = tid >> 5;
    const int wr   = wave >> 1;
    const int wc   = wave & 1;
    const int m0   = blockIdx.y * TILE_M;
    const int n0   = blockIdx.x * TILE_N;

    v8f acc[2][2] = {};

    const int ldrow = tid >> 1;          // 0..63
    const int ldk0  = (tid & 1) * 16;    // K half

    const uint32_t ldsB = (uint32_t)(uintptr_t)&sB[0][0];

    for (int k0 = 0; k0 < K; k0 += TILE_K) {
        // ---- TDM: DMA the 64x32 bf16 weight tile into LDS (wave 0 issues) ----
        if (wave == 0) {
            const uint64_t ga = (uint64_t)(uintptr_t)(Wbf + (size_t)n0 * ldw + k0);
            const uint32_t td0 = (uint32_t)(K - k0);   // remaining elems in row
            const uint32_t td1 = (uint32_t)(N - n0);   // remaining rows
            u32x4 g0;
            g0[0] = 1u;                                // count=1, user mode
            g0[1] = ldsB;                              // lds_addr
            g0[2] = (uint32_t)ga;                      // global_addr[31:0]
            g0[3] = (uint32_t)(ga >> 32) | (2u << 30); // addr[56:32] | type=2
            i32x8 g1;
            g1[0] = (int)((1u << 16) |                 // data_size = 2B
                          (1u << 20) |                 // pad_enable
                          (3u << 22) |                 // pad_interval: 16 dwords
                          (3u << 25));                 // pad_amount: 4 dwords
            g1[1] = (int)((td0 & 0xFFFFu) << 16);      // tensor_dim0[15:0]
            g1[2] = (int)(((td0 >> 16) & 0xFFFFu) | ((td1 & 0xFFFFu) << 16));
            g1[3] = (int)(((td1 >> 16) & 0xFFFFu) | (32u << 16)); // tile_dim0=32
            g1[4] = 64;                                // tile_dim1=64, tile_dim2=0
            g1[5] = ldw;                               // dim0 stride (elems)
            g1[6] = 0;
            g1[7] = 0;
            i32x4 g2 = {0, 0, 0, 0};
            i32x4 g3 = {0, 0, 0, 0};
            i32x8 g4 = {0, 0, 0, 0, 0, 0, 0, 0};
            __builtin_amdgcn_tensor_load_to_lds(g0, g1, g2, g3, g4, 0);
        }

        // ---- stage A tile ----
        if (AMODE == 0) {
            const int m = m0 + ldrow;
            const bool valid = (m < M) && (k0 + ldk0 + 16 <= K);
            const float4* ap =
                reinterpret_cast<const float4*>(A + (size_t)m * lda + k0 + ldk0);
            if (valid && k0 + TILE_K < K)
                __builtin_prefetch((const float*)ap + TILE_K, 0, 1);
            #pragma unroll
            for (int q = 0; q < 4; ++q) {
                const float4 v = valid ? ap[q] : float4{0.f, 0.f, 0.f, 0.f};
                sA[ldrow][ldk0 + 4 * q + 0] = (bf16_t)v.x;
                sA[ldrow][ldk0 + 4 * q + 1] = (bf16_t)v.y;
                sA[ldrow][ldk0 + 4 * q + 2] = (bf16_t)v.z;
                sA[ldrow][ldk0 + 4 * q + 3] = (bf16_t)v.w;
            }
        } else {
            // k -> (ci, kh, kw) table, built once per K step
            if (tid < TILE_K) {
                const int kk = k0 + tid;
                const int ci = kk / 9;
                const int r9 = kk - ci * 9;
                const int kh = r9 / 3;
                const int kw = r9 - kh * 3;
                sKbase[tid] = ci * (CONV1_H * CONV1_W) + kw;
                sKh[tid]    = kh - 2;                 // includes pad shift
            }
            __syncthreads();
            const int m  = m0 + ldrow;
            const int bt = m / CONV2_F;
            const int f  = m - bt * CONV2_F;
            const int b  = bt / TT;
            const int t  = bt - b * TT;
            const size_t bTerm = (size_t)b * 256 * CONV1_H * CONV1_W;
            #pragma unroll
            for (int e = 0; e < 16; ++e) {
                const int h  = 2 * t + sKh[ldk0 + e];
                const bool ok = (m < M) && (h >= 0) && (h < CONV1_H);
                const size_t addr =
                    ok ? (bTerm + (size_t)sKbase[ldk0 + e] + (size_t)h * CONV1_W + 2 * f)
                       : (size_t)0;
                const bf16_t v = Abf[addr];
                sA[ldrow][ldk0 + e] = ok ? v : (bf16_t)0.0f;
            }
        }

        __builtin_amdgcn_s_wait_tensorcnt(0);   // B tile resident in LDS
        __syncthreads();

        // ---- fragment gather per ISA 7.12.2 16-bit layouts ----
        const int arow = wr * 32 + (lane & 15);
        const int bcol = wc * 32 + (lane & 15);
        const int akb  = (lane >> 4) * 8;
        const int bkb  = (lane >> 4) * 16;
        v16bf afrag[2], bfrag[2];
        #pragma unroll
        for (int s = 0; s < 2; ++s) {
            #pragma unroll
            for (int i = 0; i < 16; ++i) {
                const int ka = akb + (i < 8 ? i : i + 8);
                afrag[s][i] = sA[arow + s * 16][ka];
                bfrag[s][i] = sB[bcol + s * 16][ka];
            }
        }
        #pragma unroll
        for (int mr = 0; mr < 2; ++mr)
            #pragma unroll
            for (int nc = 0; nc < 2; ++nc)
                acc[mr][nc] = __builtin_amdgcn_wmma_f32_16x16x32_bf16(
                    false, afrag[mr], false, bfrag[nc],
                    (short)0, acc[mr][nc], false, false);
        __syncthreads();
    }

    // ---- epilogue: D layout: VGPR r, lane l -> M = r + 8*(l>=16), N = l&15 ----
    #pragma unroll
    for (int mr = 0; mr < 2; ++mr) {
        #pragma unroll
        for (int nc = 0; nc < 2; ++nc) {
            const int n = n0 + wc * 32 + nc * 16 + (lane & 15);
            if (n >= N) continue;
            const float bv = bias ? bias[n] : 0.0f;
            #pragma unroll
            for (int r = 0; r < 8; ++r) {
                const int m = m0 + wr * 32 + mr * 16 + r + ((lane >> 4) * 8);
                if (m >= M) continue;
                float v = acc[mr][nc][r] + bv;
                if (ACT == 1)      v = fmaxf(v, 0.0f);
                else if (ACT == 2) v = (v > 20.0f) ? v : log1pf(__expf(v));
                else if (ACT == 3) v = v / (1.0f + __expf(-v));
                if (SMODE == 0) {
                    C[(size_t)m * ldc + n] = v;
                } else {
                    const int bt = m / CONV2_F;
                    const int f  = m - bt * CONV2_F;
                    C[(size_t)bt * EMB_K + n * CONV2_F + f] = v;
                }
            }
        }
    }
}

// ---------------------------------------------------------------------------
__global__ void cvt_bf16(const float* __restrict__ src, bf16_t* __restrict__ dst, int n)
{
    const int i = blockIdx.x * blockDim.x + threadIdx.x;
    if (i < n) dst[i] = (bf16_t)src[i];
}

// ---------------------------------------------------------------------------
// conv1: (8,1,1600,80) -> relu -> bf16 (8,256,799,39)
// ---------------------------------------------------------------------------
__global__ void conv1_relu(const float* __restrict__ x, const float* __restrict__ w,
                           const float* __restrict__ bias, bf16_t* __restrict__ out)
{
    const int idx = blockIdx.x * blockDim.x + threadIdx.x;
    const int total = BB * 256 * CONV1_H * CONV1_W;
    if (idx >= total) return;
    const int wo = idx % CONV1_W;
    const int t  = (idx / CONV1_W) % CONV1_H;
    const int co = (idx / (CONV1_W * CONV1_H)) % 256;
    const int b  = idx / (CONV1_W * CONV1_H * 256);
    float s = bias[co];
    #pragma unroll
    for (int kh = 0; kh < 3; ++kh) {
        const int hin = 2 * t + kh - 2;
        if (hin < 0 || hin >= 1600) continue;
        #pragma unroll
        for (int kw = 0; kw < 3; ++kw)
            s += x[((size_t)b * 1600 + hin) * 80 + 2 * wo + kw] * w[co * 9 + kh * 3 + kw];
    }
    out[idx] = (bf16_t)fmaxf(s, 0.0f);
}

// ---------------------------------------------------------------------------
__global__ void dwconv_silu(const float* __restrict__ xz, const float* __restrict__ cw,
                            const float* __restrict__ cb, float* __restrict__ xi)
{
    const int idx = blockIdx.x * blockDim.x + threadIdx.x;   // bt*512 + c
    if (idx >= BT * DI) return;
    const int c  = idx % DI;
    const int bt = idx / DI;
    const int b  = bt / TT;
    const int t  = bt - b * TT;
    float s = cb[c];
    #pragma unroll
    for (int j = 0; j < 4; ++j) {
        const int tt = t - 3 + j;
        if (tt >= 0) s += xz[((size_t)(b * TT + tt)) * (2 * DI) + c] * cw[c * 4 + j];
    }
    xi[idx] = s / (1.0f + __expf(-s));   // silu
}

// ---------------------------------------------------------------------------
__global__ __launch_bounds__(256)
void layernorm_k(const float* __restrict__ x, const float* __restrict__ w,
                 const float* __restrict__ bcoef, float* __restrict__ out)
{
    __shared__ float red[256];
    const int row = blockIdx.x;
    const int c   = threadIdx.x;
    const float v = x[(size_t)row * DM + c];
    red[c] = v; __syncthreads();
    for (int s = 128; s > 0; s >>= 1) { if (c < s) red[c] += red[c + s]; __syncthreads(); }
    const float mean = red[0] * (1.0f / DM);
    __syncthreads();
    const float d = v - mean;
    red[c] = d * d; __syncthreads();
    for (int s = 128; s > 0; s >>= 1) { if (c < s) red[c] += red[c + s]; __syncthreads(); }
    const float var = red[0] * (1.0f / DM);
    out[(size_t)row * DM + c] = d * rsqrtf(var + 1e-12f) * w[c] + bcoef[c];
}

// ---------------------------------------------------------------------------
__global__ void add_inplace(float* __restrict__ res, const float* __restrict__ x,
                            int n, int first)
{
    const int i = blockIdx.x * blockDim.x + threadIdx.x;
    if (i < n) res[i] = first ? x[i] : (res[i] + x[i]);
}

// ---------------------------------------------------------------------------
// selective scan: per (b, c) recurrence with state h[16]; emits y * silu(z)
// ---------------------------------------------------------------------------
__global__ __launch_bounds__(256)
void scan_kernel(const float* __restrict__ dt,    // (BT, 512)
                 const float* __restrict__ dbl,   // (BT, 48): [dt_r | B | C]
                 const float* __restrict__ u,     // xi (BT, 512)
                 const float* __restrict__ xz,    // (BT, 1024) for z
                 const float* __restrict__ A_log, // (512, 16)
                 const float* __restrict__ Dp,    // (512)
                 float* __restrict__ out)         // (BT, 512) = y * silu(z)
{
    __shared__ float sB[DS], sC[DS];
    const int b    = blockIdx.x >> 1;
    const int half = blockIdx.x & 1;
    const int c    = half * 256 + threadIdx.x;
    float Ac[DS], h[DS];
    #pragma unroll
    for (int n = 0; n < DS; ++n) { Ac[n] = -__expf(A_log[c * DS + n]); h[n] = 0.0f; }
    const float Dc = Dp[c];
    for (int t = 0; t < TT; ++t) {
        const int bt = b * TT + t;
        if (threadIdx.x < 32) {
            const int j = threadIdx.x;
            if (j < DS) sB[j]      = dbl[(size_t)bt * 48 + DTR + j];
            else        sC[j - DS] = dbl[(size_t)bt * 48 + DTR + DS + (j - DS)];
        }
        __syncthreads();
        const float dtv = dt[(size_t)bt * DI + c];
        const float uv  = u[(size_t)bt * DI + c];
        const float du  = dtv * uv;
        float y = 0.0f;
        #pragma unroll
        for (int n = 0; n < DS; ++n) {
            const float dA = __expf(dtv * Ac[n]);
            h[n] = dA * h[n] + du * sB[n];
            y += h[n] * sC[n];
        }
        y += uv * Dc;
        const float zz = xz[(size_t)bt * (2 * DI) + DI + c];
        out[(size_t)bt * DI + c] = y * (zz / (1.0f + __expf(-zz)));
        __syncthreads();
    }
}

// ---------------------------------------------------------------------------
extern "C" void kernel_launch(void* const* d_in, const int* in_sizes, int n_in,
                              void* d_out, int out_size, void* d_ws, size_t ws_size,
                              hipStream_t stream)
{
    (void)in_sizes; (void)n_in; (void)out_size; (void)ws_size;
    const float* xs_pad  = (const float*)d_in[0];
    const float* c1_w    = (const float*)d_in[2];
    const float* c1_b    = (const float*)d_in[3];
    const float* c2_w    = (const float*)d_in[4];
    const float* c2_b    = (const float*)d_in[5];
    const float* emb_w   = (const float*)d_in[6];
    const float* emb_b   = (const float*)d_in[7];
    const float* ln_w    = (const float*)d_in[8];
    const float* ln_b    = (const float*)d_in[9];
    const float* in_w    = (const float*)d_in[10];
    const float* cconv_w = (const float*)d_in[11];
    const float* cconv_b = (const float*)d_in[12];
    const float* xproj_w = (const float*)d_in[13];
    const float* dt_w    = (const float*)d_in[14];
    const float* dt_b    = (const float*)d_in[15];
    const float* A_log   = (const float*)d_in[16];
    const float* Dp      = (const float*)d_in[17];
    const float* out_w   = (const float*)d_in[18];
    const float* nf_w    = (const float*)d_in[19];
    const float* nf_b    = (const float*)d_in[20];
    const float* eo_w    = (const float*)d_in[21];
    const float* eo_b    = (const float*)d_in[22];
    const float* eol_w   = (const float*)d_in[23];
    const float* eol_b   = (const float*)d_in[24];
    float* outp = (float*)d_out;

    // ---- workspace carve (peak ~245 MB) ----
    char* ws = (char*)d_ws;
    size_t off = 0;
    auto take = [&](size_t bytes) -> char* {
        char* p = ws + off;
        off = (off + bytes + 255) & ~(size_t)255;
        return p;
    };
    bf16_t* conv1out = (bf16_t*)take((size_t)BB * 256 * CONV1_H * CONV1_W * 2);
    float* embA  = (float*)take((size_t)BT * EMB_K * 4);
    float* xbuf  = (float*)take((size_t)BT * DM * 4);
    float* resid = (float*)take((size_t)BT * DM * 4);
    float* hln   = (float*)take((size_t)BT * DM * 4);
    float* xzbuf = (float*)take((size_t)BT * 2 * DI * 4);
    float* xibuf = (float*)take((size_t)BT * DI * 4);
    float* dblb  = (float*)take((size_t)BT * 48 * 4);
    float* dtbuf = (float*)take((size_t)BT * DI * 4);
    float* ysbuf = (float*)take((size_t)BT * DI * 4);
    float* tmp2  = (float*)take((size_t)BT * DM * 4);
    // bf16 weight copies (TDM B-tile source)
    bf16_t* c2_wb    = (bf16_t*)take((size_t)256 * 2304 * 2);
    bf16_t* emb_wb   = (bf16_t*)take((size_t)256 * EMB_K * 2);
    bf16_t* in_wb    = (bf16_t*)take((size_t)6 * 2 * DI * DM * 2);
    bf16_t* xproj_wb = (bf16_t*)take((size_t)6 * 48 * DI * 2);
    bf16_t* dt_wb    = (bf16_t*)take((size_t)6 * DI * DTR * 2);
    bf16_t* out_wb   = (bf16_t*)take((size_t)6 * DM * DI * 2);
    bf16_t* eo_wb    = (bf16_t*)take((size_t)DM * DM * 2);

    auto cvt = [&](const float* s, bf16_t* d, int n) {
        cvt_bf16<<<(n + 255) / 256, 256, 0, stream>>>(s, d, n);
    };
    cvt(c2_w,    c2_wb,    256 * 2304);
    cvt(emb_w,   emb_wb,   256 * EMB_K);
    cvt(in_w,    in_wb,    6 * 2 * DI * DM);
    cvt(xproj_w, xproj_wb, 6 * 48 * DI);
    cvt(dt_w,    dt_wb,    6 * DI * DTR);
    cvt(out_w,   out_wb,   6 * DM * DI);
    cvt(eo_w,    eo_wb,    DM * DM);

    // 1) conv1 + relu -> bf16
    {
        const int total = BB * 256 * CONV1_H * CONV1_W;
        conv1_relu<<<(total + 255) / 256, 256, 0, stream>>>(xs_pad, c1_w, c1_b, conv1out);
    }
    // 2) conv2 as implicit im2col WMMA GEMM (+bias, relu), scatter to emb-A layout
    {
        const int M = BT * CONV2_F;  // 60648
        dim3 grid((256 + TILE_N - 1) / TILE_N, (M + TILE_M - 1) / TILE_M);
        gemm_wmma<1, 1, 1><<<grid, GEMM_THREADS, 0, stream>>>(
            nullptr, 0, conv1out, c2_wb, 2304, c2_b, embA, 0, M, 256, 2304);
    }
    // 3) embedding GEMM
    {
        dim3 grid(256 / TILE_N, (BT + TILE_M - 1) / TILE_M);
        gemm_wmma<0, 0, 0><<<grid, GEMM_THREADS, 0, stream>>>(
            embA, EMB_K, nullptr, emb_wb, EMB_K, emb_b, xbuf, DM, BT, 256, EMB_K);
    }

    const int NE = BT * DM;
    const int MT = (BT + TILE_M - 1) / TILE_M;   // 50
    for (int i = 0; i < 6; ++i) {
        add_inplace<<<(NE + 255) / 256, 256, 0, stream>>>(resid, xbuf, NE, i == 0 ? 1 : 0);
        layernorm_k<<<BT, 256, 0, stream>>>(resid, ln_w + i * DM, ln_b + i * DM, hln);
        {   // in_proj
            dim3 grid(1024 / TILE_N, MT);
            gemm_wmma<0, 0, 0><<<grid, GEMM_THREADS, 0, stream>>>(
                hln, DM, nullptr, in_wb + (size_t)i * 2 * DI * DM, DM,
                nullptr, xzbuf, 2 * DI, BT, 2 * DI, DM);
        }
        {   // depthwise causal conv + silu -> xi
            const int tot = BT * DI;
            dwconv_silu<<<(tot + 255) / 256, 256, 0, stream>>>(
                xzbuf, cconv_w + (size_t)i * DI * 4, cconv_b + i * DI, xibuf);
        }
        {   // x_proj
            dim3 grid(1, MT);
            gemm_wmma<0, 0, 0><<<grid, GEMM_THREADS, 0, stream>>>(
                xibuf, DI, nullptr, xproj_wb + (size_t)i * 48 * DI, DI,
                nullptr, dblb, 48, BT, 48, DI);
        }
        {   // dt: softplus(dt_r @ dt_w^T + dt_b)
            dim3 grid(DI / TILE_N, MT);
            gemm_wmma<0, 0, 2><<<grid, GEMM_THREADS, 0, stream>>>(
                dblb, 48, nullptr, dt_wb + (size_t)i * DI * DTR, DTR,
                dt_b + i * DI, dtbuf, DI, BT, DI, DTR);
        }
        scan_kernel<<<16, 256, 0, stream>>>(
            dtbuf, dblb, xibuf, xzbuf,
            A_log + (size_t)i * DI * DS, Dp + i * DI, ysbuf);
        {   // out_proj
            dim3 grid(DM / TILE_N, MT);
            gemm_wmma<0, 0, 0><<<grid, GEMM_THREADS, 0, stream>>>(
                ysbuf, DI, nullptr, out_wb + (size_t)i * DM * DI, DI,
                nullptr, xbuf, DM, BT, DM, DI);
        }
    }
    add_inplace<<<(NE + 255) / 256, 256, 0, stream>>>(resid, xbuf, NE, 0);
    layernorm_k<<<BT, 256, 0, stream>>>(resid, nf_w, nf_b, hln);
    {   // encoder output head: silu(h @ eo_w^T + eo_b)
        dim3 grid(DM / TILE_N, (BT + TILE_M - 1) / TILE_M);
        gemm_wmma<0, 0, 3><<<grid, GEMM_THREADS, 0, stream>>>(
            hln, DM, nullptr, eo_wb, DM, eo_b, tmp2, DM, BT, DM, DM);
    }
    layernorm_k<<<BT, 256, 0, stream>>>(tmp2, eol_w, eol_b, outp);
}
